// before_RPN_samfuse_27874337751547
// MI455X (gfx1250) — matrix operand order repaired and checked by reading, not compiled, not run on hardware
//
#include <hip/hip_runtime.h>
#include <hip/hip_bf16.h>

typedef __attribute__((ext_vector_type(16))) _Float16 v16h;
typedef __attribute__((ext_vector_type(8)))  _Float16 h8;
typedef __attribute__((ext_vector_type(8)))  float    v8f;

#define BATCH 2
#define SP2   576     /* 24*24 */
#define SP0   9216    /* 96*96 */

// ---------------------------------------------------------------------------
// f16 GEMM with f32 accumulation:  Y[o,p] = sum_i A(o,i)*B(i,p) (+bias) (+=Y)
// MODE 0: A row-major (OxI), B row-major (IxP)
// MODE 1: A transposed (IxO)  -> A(o,i)=Ah[i*lda+o], B row-major
// MODE 2: A row-major,        B transposed (PxI) -> B(i,p)=Bh[p*ldb+i]
// MODE 3: A row-major,        B = spatially shifted image rows (3x3 conv tap)
//
// Block = 128 threads (4 waves), output tile 64(O) x 32(P); each wave owns a
// 16x32 strip with TWO f32 accumulators, reusing its A fragment across two
// v_wmma_f32_16x16x32_f16 per K-slab (8 WMMA / 6KB staged per block-slab).
// Double-buffered LDS staging: slab k+32 is loaded (b128) while slab k feeds
// the matrix pipe; ONE barrier per slab. O multiple of 64, P multiple of 32.
// No divergent branches around WMMA: EXEC all-1s.
// ---------------------------------------------------------------------------
template<int MODE>
__global__ void __launch_bounds__(128)
gemm_f16(const _Float16* __restrict__ Ah, int lda,
         const _Float16* __restrict__ Bh, int ldb,
         const float* __restrict__ bias,
         float* __restrict__ Y, _Float16* __restrict__ Yh,
         int O, int P, int I,
         int Hd, int Wd, int dy, int dx, int accumulate)
{
  constexpr bool AT = (MODE == 1);
  constexpr int  BM = (MODE == 2) ? 1 : (MODE == 3) ? 2 : 0;

  __shared__ __attribute__((aligned(16))) _Float16 As[2][64 * 40];
  __shared__ __attribute__((aligned(16))) _Float16 Bs[2][32 * 40];

  const int tid  = threadIdx.x;
  const int lane = tid & 31;
  const int wave = tid >> 5;
  const int o0g = blockIdx.y * 64, p0g = blockIdx.x * 32;

  // fragment addressing (wave32 WMMA layouts)
  const int mArow = wave * 16 + (lane & 15);  // A rows: lanes 0-15 K=0..15, 16-31 K=16..31
  const int koffA = (lane >> 4) * 16;
  const int pBrow = lane & 15;                // B: lanes 0-15 K=0..15, 16-31 K=16..31
  const int kbB   = (lane >> 4) * 16;

  // staging roles: 128 threads x 8 halves per pass
  const int r4 = tid >> 2, s4 = tid & 3;

  auto stage = [&](int kk, int buf) {
    // ---- A tile: 64 o x 32 k (two passes of 8 halves/thread) ----
    if (!AT) {
#pragma unroll
      for (int rep = 0; rep < 2; ++rep) {
        int r = rep * 32 + r4;
        h8 v = *(const h8*)(Ah + (size_t)(o0g + r) * lda + kk + s4 * 8);
        *(h8*)(&As[buf][r * 40 + s4 * 8]) = v;
      }
      if (kk + 32 < I)
        __builtin_prefetch(Ah + (size_t)(o0g + r4) * lda + kk + 32 + s4 * 8, 0, 1);
    } else {
#pragma unroll
      for (int rep = 0; rep < 2; ++rep) {
        int oseg = s4 * 8 + rep * 32;
        h8 v = *(const h8*)(Ah + (size_t)(kk + r4) * lda + o0g + oseg);
#pragma unroll
        for (int j = 0; j < 8; ++j) As[buf][(oseg + j) * 40 + r4] = v[j];
      }
    }
    // ---- B tile into p-major LDS: 32 p x 32 k ----
    if (BM == 0) {            // row-major IxP: rows contiguous along p
      h8 v = *(const h8*)(Bh + (size_t)(kk + r4) * ldb + p0g + s4 * 8);
#pragma unroll
      for (int j = 0; j < 8; ++j) Bs[buf][(s4 * 8 + j) * 40 + r4] = v[j];
    } else if (BM == 1) {     // PxI: rows contiguous along k
      h8 v = *(const h8*)(Bh + (size_t)(p0g + r4) * ldb + kk + s4 * 8);
      *(h8*)(&Bs[buf][r4 * 40 + s4 * 8]) = v;
    } else {                  // shifted conv tap with zero padding
#pragma unroll
      for (int j = 0; j < 8; ++j) {
        int pl = s4 * 8 + j;
        int p  = p0g + pl;
        int yy = p / Wd + dy, xx = p % Wd + dx;
        _Float16 v = (_Float16)0.f;
        if (yy >= 0 && yy < Hd && xx >= 0 && xx < Wd)
          v = Bh[(size_t)(kk + r4) * ldb + yy * Wd + xx];
        Bs[buf][pl * 40 + r4] = v;
      }
    }
  };

  v8f acc0 = {0.f,0.f,0.f,0.f,0.f,0.f,0.f,0.f};
  v8f acc1 = {0.f,0.f,0.f,0.f,0.f,0.f,0.f,0.f};

  stage(0, 0);
  int buf = 0;
  for (int k = 0; k < I; k += 32) {
    __syncthreads();
    if (k + 32 < I) stage(k + 32, buf ^ 1);   // overlap next slab with compute
    const _Float16* Ab = &As[buf][0];
    const _Float16* Bb = &Bs[buf][0];
    v16h a, b0, b1;
    {
      h8 lo = *(const h8*)(Ab + mArow * 40 + koffA);
      h8 hi = *(const h8*)(Ab + mArow * 40 + koffA + 8);
#pragma unroll
      for (int j = 0; j < 8; ++j) { a[j] = lo[j]; a[8 + j] = hi[j]; }
    }
    {
      h8 lo = *(const h8*)(Bb + pBrow * 40 + kbB);
      h8 hi = *(const h8*)(Bb + pBrow * 40 + kbB + 8);
#pragma unroll
      for (int j = 0; j < 8; ++j) { b0[j] = lo[j]; b0[8 + j] = hi[j]; }
    }
    {
      h8 lo = *(const h8*)(Bb + (pBrow + 16) * 40 + kbB);
      h8 hi = *(const h8*)(Bb + (pBrow + 16) * 40 + kbB + 8);
#pragma unroll
      for (int j = 0; j < 8; ++j) { b1[j] = lo[j]; b1[8 + j] = hi[j]; }
    }
    acc0 = __builtin_amdgcn_wmma_f32_16x16x32_f16(false, a, false, b0,
                                                  (short)0, acc0, false, false);
    acc1 = __builtin_amdgcn_wmma_f32_16x16x32_f16(false, a, false, b1,
                                                  (short)0, acc1, false, false);
    buf ^= 1;
  }

  // ---- epilogue: C/D layout rows m = base + v (+8 for hi half-wave) ----
  const int n0 = p0g + (lane & 15);
  const int mb = o0g + wave * 16 + ((lane >> 4) * 8);
#pragma unroll
  for (int v = 0; v < 8; ++v) {
    int m = mb + v;
    float bv = bias ? bias[m] : 0.f;
    size_t i0 = (size_t)m * P + n0;
    size_t i1 = i0 + 16;
    float v0 = acc0[v] + bv;
    float v1 = acc1[v] + bv;
    if (accumulate) { v0 += Y[i0]; v1 += Y[i1]; }
    Y[i0] = v0; Y[i1] = v1;
    if (Yh) { Yh[i0] = (_Float16)v0; Yh[i1] = (_Float16)v1; }
  }
}

// ---------------------------------------------------------------------------
// elementwise / resize kernels
// ---------------------------------------------------------------------------
__global__ void cvt16(_Float16* __restrict__ dst, const float* __restrict__ src, int n)
{
  int i = blockIdx.x * blockDim.x + threadIdx.x;
  if (i < n) dst[i] = (_Float16)src[i];
}

// w_fusion (384,768,3,3) -> tap-major f16: dst[tap][o][c]
__global__ void repack_wfus(_Float16* __restrict__ dst, const float* __restrict__ src)
{
  int i = blockIdx.x * blockDim.x + threadIdx.x;
  const int n = 9 * 768 * 384;
  if (i >= n) return;
  int tap = i / (768 * 384);
  int rem = i % (768 * 384);
  int o = rem / 768, c = rem % 768;
  dst[i] = (_Float16)src[(size_t)(o * 768 + c) * 9 + tap];
}

__global__ void fuse5(const float* __restrict__ f0, const float* __restrict__ f1,
                      const float* __restrict__ f2, const float* __restrict__ f3,
                      const float* __restrict__ f4,
                      float* __restrict__ out, _Float16* __restrict__ outh,
                      int BC, int Ho)
{
  int idx = blockIdx.x * blockDim.x + threadIdx.x;
  int total = BC * Ho * Ho;
  if (idx >= total) return;
  int x  = idx % Ho;
  int y  = (idx / Ho) % Ho;
  int bc = idx / (Ho * Ho);
  const int hs[5] = {96, 48, 24, 12, 6};
  const float* fp[5] = {f0, f1, f2, f3, f4};
  float accv = 0.f;
#pragma unroll
  for (int l = 0; l < 5; ++l) {
    int H = hs[l];
    const float* src = fp[l] + (size_t)bc * H * H;
    if (H == Ho) {
      accv += src[y * H + x];
    } else {
      float sy = (float)y * (float)(H - 1) / (float)(Ho - 1);
      float sx = (float)x * (float)(H - 1) / (float)(Ho - 1);
      int y0 = (int)floorf(sy), x0 = (int)floorf(sx);
      int y1 = min(y0 + 1, H - 1), x1 = min(x0 + 1, H - 1);
      float wy = sy - (float)y0, wx = sx - (float)x0;
      accv += src[y0 * H + x0] * (1.f - wy) * (1.f - wx)
            + src[y0 * H + x1] * (1.f - wy) * wx
            + src[y1 * H + x0] * wy * (1.f - wx)
            + src[y1 * H + x1] * wy * wx;
    }
  }
  float v = accv * 0.2f;
  out[idx] = v;
  if (outh) outh[idx] = (_Float16)v;
}

__global__ void restore_add(const float* __restrict__ res, int Hr,
                            const float* __restrict__ feat,
                            float* __restrict__ out, int BC, int Hl)
{
  int idx = blockIdx.x * blockDim.x + threadIdx.x;
  int total = BC * Hl * Hl;
  if (idx >= total) return;
  int x  = idx % Hl;
  int y  = (idx / Hl) % Hl;
  int bc = idx / (Hl * Hl);
  const float* src = res + (size_t)bc * Hr * Hr;
  float v;
  if (Hr == Hl) {
    v = src[y * Hr + x];
  } else {
    float sy = (float)y * (float)(Hr - 1) / (float)(Hl - 1);
    float sx = (float)x * (float)(Hr - 1) / (float)(Hl - 1);
    int y0 = (int)floorf(sy), x0 = (int)floorf(sx);
    int y1 = min(y0 + 1, Hr - 1), x1 = min(x0 + 1, Hr - 1);
    float wy = sy - (float)y0, wx = sx - (float)x0;
    v = src[y0 * Hr + x0] * (1.f - wy) * (1.f - wx)
      + src[y0 * Hr + x1] * (1.f - wy) * wx
      + src[y1 * Hr + x0] * wy * (1.f - wx)
      + src[y1 * Hr + x1] * wy * wx;
  }
  out[idx] = v + feat[idx];
}

__global__ void softmax_rows(float* __restrict__ A, _Float16* __restrict__ Ah, int Q)
{
  float*    row  = A  + (size_t)blockIdx.x * Q;
  _Float16* rowh = Ah + (size_t)blockIdx.x * Q;
  __shared__ float red[256];
  int t = threadIdx.x;
  float mx = -3.4e38f;
  for (int q = t; q < Q; q += 256) mx = fmaxf(mx, row[q]);
  red[t] = mx; __syncthreads();
  for (int s = 128; s > 0; s >>= 1) {
    if (t < s) red[t] = fmaxf(red[t], red[t + s]);
    __syncthreads();
  }
  mx = red[0]; __syncthreads();
  float sum = 0.f;
  for (int q = t; q < Q; q += 256) { float v = expf(row[q] - mx); row[q] = v; sum += v; }
  red[t] = sum; __syncthreads();
  for (int s = 128; s > 0; s >>= 1) {
    if (t < s) red[t] += red[t + s];
    __syncthreads();
  }
  float inv = 1.f / red[0];
  for (int q = t; q < Q; q += 256) {
    float v = row[q] * inv;
    row[q] = v;
    rowh[q] = (_Float16)v;
  }
}

// mask[e] = sigmoid(sum_c att[c,e]*gate[c]); att[:,e] *= mask[e]  (C=384)
__global__ void gate_mask(float* __restrict__ att, _Float16* __restrict__ atth,
                          const float* __restrict__ gate, int E)
{
  int e = blockIdx.x * blockDim.x + threadIdx.x;
  if (e >= E) return;
  float s = 0.f;
  for (int c = 0; c < 384; ++c) s += att[(size_t)c * E + e] * gate[c];
  float m = 1.f / (1.f + expf(-s));
  for (int c = 0; c < 384; ++c) {
    size_t i = (size_t)c * E + e;
    float v = att[i] * m;
    att[i] = v;
    atth[i] = (_Float16)v;
  }
}

__global__ void gru_prevr(_Float16* __restrict__ ph, const float* __restrict__ prev,
                          const float* __restrict__ Rp, int n)
{
  int i = blockIdx.x * blockDim.x + threadIdx.x;
  if (i < n) ph[i] = (_Float16)(prev[i] * (1.f / (1.f + expf(-Rp[i]))));
}

__global__ void gru_combine(float* __restrict__ state, _Float16* __restrict__ stateh,
                            const float* __restrict__ Up, const float* __restrict__ Op, int n)
{
  int i = blockIdx.x * blockDim.x + threadIdx.x;
  if (i < n) {
    float u = 1.f / (1.f + expf(-Up[i]));
    float o = tanhf(Op[i]);
    float v = state[i] * (1.f - u) + o * u;
    state[i] = v;
    stateh[i] = (_Float16)v;
  }
}

// ---------------------------------------------------------------------------
// host helpers
// ---------------------------------------------------------------------------
template<int MODE>
static inline void launch_gemm(hipStream_t st, const _Float16* A, int lda,
                               const _Float16* B, int ldb, const float* bias,
                               float* Y, _Float16* Yh, int O, int P, int I,
                               int acc, int H = 0, int dy = 0, int dx = 0)
{
  dim3 g(P / 32, O / 64), b(128);
  gemm_f16<MODE><<<g, b, 0, st>>>(A, lda, B, ldb, bias, Y, Yh, O, P, I, H, H, dy, dx, acc);
}

static void gen_att(hipStream_t st, const _Float16* ex_h, int E,
                    const _Float16* q_h, int Q,
                    float* att, _Float16* att_h,
                    float* exc, _Float16* exc_h,
                    float* Amat, _Float16* Amat_h,
                    const _Float16* wlin_h, const float* gate)
{
  launch_gemm<0>(st, wlin_h, 384, ex_h, E, nullptr, exc, exc_h, 384, E, 384, 0);
  launch_gemm<1>(st, exc_h, E, q_h, Q, nullptr, Amat, nullptr, E, Q, 384, 0);
  softmax_rows<<<E, 256, 0, st>>>(Amat, Amat_h, Q);
  launch_gemm<2>(st, q_h, Q, Amat_h, Q, nullptr, att, nullptr, 384, E, Q, 0);
  gate_mask<<<(E + 255) / 256, 256, 0, st>>>(att, att_h, gate, E);
}

static void fusion_conv(hipStream_t st, const _Float16* wfusT_h, const float* bfus,
                        const _Float16* ah, const _Float16* bh,
                        float* y, _Float16* yh, int H)
{
  int P = H * H, idx = 0;
  for (int part = 0; part < 2; ++part) {
    const _Float16* X = part ? bh : ah;
    for (int ky = 0; ky < 3; ++ky)
      for (int kx = 0; kx < 3; ++kx) {
        const _Float16* Wp = wfusT_h + (size_t)(ky * 3 + kx) * 768 * 384 + part * 384;
        launch_gemm<3>(st, Wp, 768, X, P, (idx == 0) ? bfus : nullptr,
                       y, (idx == 17) ? yh : nullptr, 384, P, 384,
                       (idx == 0) ? 0 : 1, H, ky - 1, kx - 1);
        ++idx;
      }
  }
}

static void gru_step(hipStream_t st, const _Float16* inp_h,
                     float* state, _Float16* state_h, int E,
                     const _Float16* wu_h, const float* bu,
                     const _Float16* wr_h, const float* br,
                     const _Float16* wo_h, const float* bo,
                     float* U, float* R, float* Og, _Float16* prevr_h)
{
  int n = 384 * E;
  launch_gemm<0>(st, wu_h,       768, inp_h,   E, bu,      U,  nullptr, 384, E, 384, 0);
  launch_gemm<0>(st, wu_h + 384, 768, state_h, E, nullptr, U,  nullptr, 384, E, 384, 1);
  launch_gemm<0>(st, wr_h,       768, inp_h,   E, br,      R,  nullptr, 384, E, 384, 0);
  launch_gemm<0>(st, wr_h + 384, 768, state_h, E, nullptr, R,  nullptr, 384, E, 384, 1);
  gru_prevr<<<(n + 255) / 256, 256, 0, st>>>(prevr_h, state, R, n);
  launch_gemm<0>(st, wo_h,       768, inp_h,   E, bo,      Og, nullptr, 384, E, 384, 0);
  launch_gemm<0>(st, wo_h + 384, 768, prevr_h, E, nullptr, Og, nullptr, 384, E, 384, 1);
  gru_combine<<<(n + 255) / 256, 256, 0, st>>>(state, state_h, U, Og, n);
}

// ---------------------------------------------------------------------------
extern "C" void kernel_launch(void* const* d_in, const int* in_sizes, int n_in,
                              void* d_out, int out_size, void* d_ws, size_t ws_size,
                              hipStream_t stream)
{
  (void)in_sizes; (void)n_in; (void)out_size; (void)ws_size;

  const float* tarL[5]; const float* refL[5]; const float* matL[5];
  for (int l = 0; l < 5; ++l) {
    tarL[l] = (const float*)d_in[l];
    refL[l] = (const float*)d_in[5 + l];
    matL[l] = (const float*)d_in[10 + l];
  }
  const float* sam      = (const float*)d_in[15];
  const float* w_conv1  = (const float*)d_in[16];
  const float* b_conv1  = (const float*)d_in[17];
  const float* w_conv2  = (const float*)d_in[18];
  const float* b_conv2  = (const float*)d_in[19];
  const float* w_fuse   = (const float*)d_in[20];
  const float* b_fuse   = (const float*)d_in[21];
  const float* w_lin    = (const float*)d_in[22];
  const float* w_gate   = (const float*)d_in[23];
  const float* w_fusion = (const float*)d_in[24];
  const float* b_fusion = (const float*)d_in[25];
  const float* w_r = (const float*)d_in[26]; const float* b_r = (const float*)d_in[27];
  const float* w_u = (const float*)d_in[28]; const float* b_u = (const float*)d_in[29];
  const float* w_o = (const float*)d_in[30]; const float* b_o = (const float*)d_in[31];

  // -------- f32 arena --------
  float* ws = (float*)d_ws;
  size_t off = 0;
  auto arena = [&](size_t n) { float* p = ws + off; off += n; return p; };
  float* refs = arena((size_t)BATCH * 384 * SP0);
  float* tars = arena((size_t)BATCH * 384 * SP2);
  float* mats = arena((size_t)BATCH * 384 * SP2);
  float* reff = arena((size_t)BATCH * 256 * SP0);
  float* tarf = arena((size_t)BATCH * 256 * SP2);
  float* tmp  = arena((size_t)BATCH * 256 * SP2);
  float* exc  = arena((size_t)384 * SP0);          // ex_corr / O-gate / tar conv2 out
  float* Amat = arena((size_t)SP0 * SP2);          // attention / ref conv2 out
  float* atta = arena((size_t)384 * SP0);          // att pair a / U
  float* attb = arena((size_t)384 * SP0);          // att pair b / R
  float* att1 = arena((size_t)384 * SP0);
  float* att2 = arena((size_t)384 * SP2);
  float* att3 = arena((size_t)384 * SP2);

  // -------- f16 arena (16B-aligned: all sizes multiples of 8) --------
  off = (off + 7) & ~(size_t)7;
  _Float16* hbase = (_Float16*)(ws + off);
  size_t hoff = 0;
  auto harena = [&](size_t n) { _Float16* p = hbase + hoff; hoff += n; return p; };
  _Float16* wfuse_h  = harena(256 * 512);
  _Float16* wconv1_h = harena(384 * 256);
  _Float16* wconv2_h = harena(256 * 384);
  _Float16* wlin_h   = harena(384 * 384);
  _Float16* wu_h     = harena(384 * 768);
  _Float16* wr_h     = harena(384 * 768);
  _Float16* wo_h     = harena(384 * 768);
  _Float16* wfusT_h  = harena((size_t)9 * 768 * 384);
  _Float16* sam_h    = harena((size_t)BATCH * 256 * SP2);
  _Float16* tarf_h   = harena((size_t)BATCH * 256 * SP2);
  _Float16* reff_h   = harena((size_t)BATCH * 256 * SP0);
  _Float16* tmp_h    = harena((size_t)BATCH * 256 * SP2);
  _Float16* refs_h   = harena((size_t)BATCH * 384 * SP0);
  _Float16* tars_h   = harena((size_t)BATCH * 384 * SP2);
  _Float16* mats_h   = harena((size_t)BATCH * 384 * SP2);
  _Float16* exc_h    = harena((size_t)384 * SP0);
  _Float16* Amat_h   = harena((size_t)SP0 * SP2);
  _Float16* atta_h   = harena((size_t)384 * SP0);
  _Float16* attb_h   = harena((size_t)384 * SP0);
  _Float16* att1_h   = harena((size_t)384 * SP0);
  _Float16* att2_h   = harena((size_t)384 * SP2);
  _Float16* att3_h   = harena((size_t)384 * SP2);
  _Float16* prevr_h  = harena((size_t)384 * SP0);

  // -------- phase 0: weight conversion / repack --------
  auto cvt = [&](_Float16* d, const float* s, size_t n) {
    cvt16<<<((n + 255) / 256), 256, 0, stream>>>(d, s, (int)n);
  };
  cvt(wfuse_h,  w_fuse,  256 * 512);
  cvt(wconv1_h, w_conv1, 384 * 256);
  cvt(wconv2_h, w_conv2, 256 * 384);
  cvt(wlin_h,   w_lin,   384 * 384);
  cvt(wu_h,     w_u,     384 * 768);
  cvt(wr_h,     w_r,     384 * 768);
  cvt(wo_h,     w_o,     384 * 768);
  cvt(sam_h,    sam,     (size_t)BATCH * 256 * SP2);
  {
    size_t n = (size_t)9 * 768 * 384;
    repack_wfus<<<(n + 255) / 256, 256, 0, stream>>>(wfusT_h, w_fusion);
  }

  // -------- phase 1: multiscale fuse + input projections --------
  {
    size_t n;
    n = (size_t)BATCH * 256 * SP2;
    fuse5<<<(n + 255) / 256, 256, 0, stream>>>(tarL[0], tarL[1], tarL[2], tarL[3], tarL[4],
                                               tarf, tarf_h, BATCH * 256, 24);
    n = (size_t)BATCH * 256 * SP0;
    fuse5<<<(n + 255) / 256, 256, 0, stream>>>(refL[0], refL[1], refL[2], refL[3], refL[4],
                                               reff, reff_h, BATCH * 256, 96);
    n = (size_t)BATCH * 384 * SP2;
    fuse5<<<(n + 255) / 256, 256, 0, stream>>>(matL[0], matL[1], matL[2], matL[3], matL[4],
                                               mats, mats_h, BATCH * 384, 24);
  }
  for (int b = 0; b < BATCH; ++b) {
    size_t o256s = (size_t)b * 256 * SP2;
    launch_gemm<0>(stream, wfuse_h,       512, tarf_h + o256s, SP2, b_fuse,
                   tmp + o256s, nullptr,        256, SP2, 256, 0);
    launch_gemm<0>(stream, wfuse_h + 256, 512, sam_h + o256s,  SP2, nullptr,
                   tmp + o256s, tmp_h + o256s,  256, SP2, 256, 1);
    launch_gemm<0>(stream, wconv1_h, 256, tmp_h + o256s, SP2, b_conv1,
                   tars + (size_t)b * 384 * SP2, tars_h + (size_t)b * 384 * SP2,
                   384, SP2, 256, 0);
    launch_gemm<0>(stream, wconv1_h, 256, reff_h + (size_t)b * 256 * SP0, SP0, b_conv1,
                   refs + (size_t)b * 384 * SP0, refs_h + (size_t)b * 384 * SP0,
                   384, SP0, 256, 0);
  }

  // -------- phase 2: iterative cross-attention + GRU (per sample) --------
  for (int b = 0; b < BATCH; ++b) {
    float*     R_  = refs   + (size_t)b * 384 * SP0;
    _Float16*  Rh  = refs_h + (size_t)b * 384 * SP0;
    float*     T_  = tars   + (size_t)b * 384 * SP2;
    _Float16*  Th  = tars_h + (size_t)b * 384 * SP2;
    float*     M_  = mats   + (size_t)b * 384 * SP2;
    _Float16*  Mh  = mats_h + (size_t)b * 384 * SP2;
    for (int it = 0; it < 2; ++it) {
      gen_att(stream, Rh, SP0, Th, SP2, atta, atta_h, exc, exc_h, Amat, Amat_h, wlin_h, w_gate);
      gen_att(stream, Rh, SP0, Mh, SP2, attb, attb_h, exc, exc_h, Amat, Amat_h, wlin_h, w_gate);
      fusion_conv(stream, wfusT_h, b_fusion, atta_h, attb_h, att1, att1_h, 96);

      gen_att(stream, Th, SP2, Rh, SP0, atta, atta_h, exc, exc_h, Amat, Amat_h, wlin_h, w_gate);
      gen_att(stream, Th, SP2, Mh, SP2, attb, attb_h, exc, exc_h, Amat, Amat_h, wlin_h, w_gate);
      fusion_conv(stream, wfusT_h, b_fusion, atta_h, attb_h, att2, att2_h, 24);

      gen_att(stream, Mh, SP2, Rh, SP0, atta, atta_h, exc, exc_h, Amat, Amat_h, wlin_h, w_gate);
      gen_att(stream, Mh, SP2, Th, SP2, attb, attb_h, exc, exc_h, Amat, Amat_h, wlin_h, w_gate);
      fusion_conv(stream, wfusT_h, b_fusion, atta_h, attb_h, att3, att3_h, 24);

      gru_step(stream, att1_h, R_, Rh, SP0, wu_h, b_u, wr_h, b_r, wo_h, b_o,
               atta, attb, exc, prevr_h);
      gru_step(stream, att2_h, T_, Th, SP2, wu_h, b_u, wr_h, b_r, wo_h, b_o,
               atta, attb, exc, prevr_h);
      gru_step(stream, att3_h, M_, Mh, SP2, wu_h, b_u, wr_h, b_r, wo_h, b_o,
               atta, attb, exc, prevr_h);
    }
  }

  // -------- phase 3: output projections + restore to all levels --------
  float* cvt_t = exc;   // (B,256,24,24)
  float* cvt_r = Amat;  // (B,256,96,96)
  for (int b = 0; b < BATCH; ++b) {
    launch_gemm<0>(stream, wconv2_h, 384, tars_h + (size_t)b * 384 * SP2, SP2, b_conv2,
                   cvt_t + (size_t)b * 256 * SP2, nullptr, 256, SP2, 384, 0);
    launch_gemm<0>(stream, wconv2_h, 384, refs_h + (size_t)b * 384 * SP0, SP0, b_conv2,
                   cvt_r + (size_t)b * 256 * SP0, nullptr, 256, SP0, 384, 0);
  }

  const int hs[5] = {96, 48, 24, 12, 6};
  float* out = (float*)d_out;
  size_t oo = 0;
  for (int l = 0; l < 5; ++l) {  // new_match
    int H = hs[l]; size_t n = (size_t)BATCH * 384 * H * H;
    restore_add<<<(n + 255) / 256, 256, 0, stream>>>(mats, 24, matL[l], out + oo,
                                                     BATCH * 384, H);
    oo += n;
  }
  for (int l = 0; l < 5; ++l) {  // new_tar
    int H = hs[l]; size_t n = (size_t)BATCH * 256 * H * H;
    restore_add<<<(n + 255) / 256, 256, 0, stream>>>(cvt_t, 24, tarL[l], out + oo,
                                                     BATCH * 256, H);
    oo += n;
  }
  for (int l = 0; l < 5; ++l) {  // new_ref
    int H = hs[l]; size_t n = (size_t)BATCH * 256 * H * H;
    restore_add<<<(n + 255) / 256, 256, 0, stream>>>(cvt_r, 96, refL[l], out + oo,
                                                     BATCH * 256, H);
    oo += n;
  }
}
